// SDPAttention_3796751089839
// MI455X (gfx1250) — matrix operand order repaired and checked by reading, not compiled, not run on hardware
//
#include <hip/hip_runtime.h>
#include <hip/hip_bf16.h>

typedef __attribute__((ext_vector_type(16))) __bf16 v16bf;
typedef __attribute__((ext_vector_type(8)))  __bf16 v8bf;
typedef __attribute__((ext_vector_type(4)))  __bf16 v4bf;
typedef __attribute__((ext_vector_type(8)))  float  v8f;

#define T_SEQ 4096
#define DHEAD 64
#define KT 64                 // keys per tile (4 x 16-col WMMA sub-tiles)
#define TQ 16                 // query rows per wave
#define NWAVES 8
#define BLOCK_Q (NWAVES*TQ)   // 128

// LDS row strides in bf16 elements; 144B rows -> row*36 mod 64 banks, all distinct
#define KSTR 72
#define VSTR 72
#define PSTR 72

__device__ __forceinline__ __bf16 f2bf(float f) { return (__bf16)f; }

__device__ __forceinline__ v16bf cat8(v8bf lo, v8bf hi) {
  return __builtin_shufflevector(lo, hi, 0,1,2,3,4,5,6,7,8,9,10,11,12,13,14,15);
}

// xor-shuffle within 16-lane halves via ds_swizzle imm (and_mask=0x1f, or=0, xor=m)
#define SWZ(x, m) __builtin_bit_cast(float, \
    __builtin_amdgcn_ds_swizzle(__builtin_bit_cast(int, (x)), 0x7C00 | (m)))

__device__ __forceinline__ float rmax16(float x) {
  x = fmaxf(x, SWZ(x, 1)); x = fmaxf(x, SWZ(x, 2));
  x = fmaxf(x, SWZ(x, 4)); x = fmaxf(x, SWZ(x, 8));
  return x;
}
__device__ __forceinline__ float radd16(float x) {
  x += SWZ(x, 1); x += SWZ(x, 2); x += SWZ(x, 4); x += SWZ(x, 8);
  return x;
}

__global__ __launch_bounds__(256, 2)
void fa_causal_fwd(const float* __restrict__ Q, const float* __restrict__ K,
                   const float* __restrict__ V, float* __restrict__ O) {
  __shared__ __bf16 lk[KT * KSTR];            // K tile row-major [key][d]
  __shared__ __bf16 lv[DHEAD * VSTR];         // V tile transposed [d][key]
  __shared__ __bf16 lp[NWAVES * 16 * PSTR];   // per-wave P scratch [row][key]

  const int tid  = threadIdx.x;
  const int lane = tid & 31;                  // wave32
  const int w    = tid >> 5;
  const int l16  = lane & 15;
  const int hi   = lane >> 4;
  const int b    = blockIdx.y;
  const int qbase = blockIdx.x * BLOCK_Q;
  const int qwave = qbase + w * TQ;

  const float* Qb = Q + (size_t)b * T_SEQ * DHEAD;
  const float* Kb = K + (size_t)b * T_SEQ * DHEAD;
  const float* Vb = V + (size_t)b * T_SEQ * DHEAD;
  float*       Ob = O + (size_t)b * T_SEQ * DHEAD;

  // ---- Q rows -> WMMA A-layout (bf16), 1/sqrt(d) folded in ----
  v16bf qa[2];
  {
    const float* qr = Qb + (size_t)(qwave + l16) * DHEAD;
    const int s = hi * 8;
#pragma unroll
    for (int c = 0; c < 2; ++c) {
      float t[16];
      *(float4*)&t[0]  = *(const float4*)(qr + 32*c + s);
      *(float4*)&t[4]  = *(const float4*)(qr + 32*c + s + 4);
      *(float4*)&t[8]  = *(const float4*)(qr + 32*c + 16 + s);
      *(float4*)&t[12] = *(const float4*)(qr + 32*c + 16 + s + 4);
#pragma unroll
      for (int h = 0; h < 16; ++h) qa[c][h] = f2bf(t[h] * 0.125f);
    }
  }

  v8f acc[4] = {v8f{}, v8f{}, v8f{}, v8f{}};
  float m[8], l[8];
#pragma unroll
  for (int r = 0; r < 8; ++r) { m[r] = -1e30f; l[r] = 0.0f; }

  const int nkt = (qbase + BLOCK_Q) / KT;
  for (int kt = 0; kt < nkt; ++kt) {
    const int kb = kt * KT;
    __syncthreads();
    // ---- stage K (packed b64 stores) and V^T (b16 scatter), f32->bf16 ----
#pragma unroll
    for (int i = 0; i < (KT * DHEAD) / (256 * 4); ++i) {   // 4 iters, float4 each
      int idx4 = (i * 256 + tid) * 4;
      int r = idx4 >> 6, c = idx4 & 63;
      float4 kf = *(const float4*)(Kb + (size_t)(kb + r) * DHEAD + c);
      float4 vf = *(const float4*)(Vb + (size_t)(kb + r) * DHEAD + c);
      v4bf kp = { f2bf(kf.x), f2bf(kf.y), f2bf(kf.z), f2bf(kf.w) };
      *(v4bf*)&lk[r * KSTR + c] = kp;                      // rows contiguous
      lv[(c + 0) * VSTR + r] = f2bf(vf.x);                 // transpose scatter
      lv[(c + 1) * VSTR + r] = f2bf(vf.y);
      lv[(c + 2) * VSTR + r] = f2bf(vf.z);
      lv[(c + 3) * VSTR + r] = f2bf(vf.w);
    }
    __syncthreads();

    if (kb <= qwave + (TQ - 1)) {   // wave-uniform: EXEC all-1s inside (WMMA req)
      // ---- S = (Q*scale) K^T : four 16x16 tiles over this 64-key slab ----
      v8f s[4] = {v8f{}, v8f{}, v8f{}, v8f{}};
#pragma unroll
      for (int ns = 0; ns < 4; ++ns)
#pragma unroll
        for (int c = 0; c < 2; ++c) {
          int ko = (ns * 16 + l16) * KSTR + 32 * c + hi * 8;
          v16bf bk = cat8(*(const v8bf*)&lk[ko], *(const v8bf*)&lk[ko + 16]);
          s[ns] = __builtin_amdgcn_wmma_f32_16x16x32_bf16(false, qa[c], false, bk,
                                                          (short)0, s[ns], false, false);
        }

      float p[4][8];
#pragma unroll
      for (int ns = 0; ns < 4; ++ns)
#pragma unroll
        for (int r = 0; r < 8; ++r) p[ns][r] = s[ns][r];

      // ---- causal mask: only on diagonal-crossing tiles (wave-uniform) ----
      if (kb + KT - 1 > qwave) {
#pragma unroll
        for (int ns = 0; ns < 4; ++ns) {
          int col = kb + ns * 16 + l16;
#pragma unroll
          for (int r = 0; r < 8; ++r) {
            int qr = qwave + r + hi * 8;
            if (col > qr) p[ns][r] = -1e30f;
          }
        }
      }

      // ---- online softmax ----
      float alpha[8], rs[8];
#pragma unroll
      for (int r = 0; r < 8; ++r) {
        float rm = fmaxf(fmaxf(p[0][r], p[1][r]), fmaxf(p[2][r], p[3][r]));
        rm = rmax16(rm);
        float mn = fmaxf(m[r], rm);
        alpha[r] = __expf(m[r] - mn);
        m[r] = mn;
#pragma unroll
        for (int ns = 0; ns < 4; ++ns) p[ns][r] = __expf(p[ns][r] - mn);
        rs[r] = radd16((p[0][r] + p[1][r]) + (p[2][r] + p[3][r]));
      }
#pragma unroll
      for (int r = 0; r < 8; ++r) {
        l[r] = l[r] * alpha[r] + rs[r];
        acc[0][r] *= alpha[r]; acc[1][r] *= alpha[r];
        acc[2][r] *= alpha[r]; acc[3][r] *= alpha[r];
      }

      // ---- P: C/D layout -> A layout via per-wave LDS scratch ----
      __bf16* pw = &lp[w * 16 * PSTR];
#pragma unroll
      for (int r = 0; r < 8; ++r) {
        int row = r + hi * 8;
#pragma unroll
        for (int ns = 0; ns < 4; ++ns)
          pw[row * PSTR + ns * 16 + l16] = f2bf(p[ns][r]);
      }
      asm volatile("s_wait_dscnt 0" ::: "memory");   // same-wave LDS RAW
      v16bf pa[2];
#pragma unroll
      for (int c = 0; c < 2; ++c) {
        int ap = w * 16 * PSTR + l16 * PSTR + 32 * c + hi * 8;
        pa[c] = cat8(*(const v8bf*)&lp[ap], *(const v8bf*)&lp[ap + 16]);
      }

      // ---- O += P V ----
#pragma unroll
      for (int dt = 0; dt < 4; ++dt)
#pragma unroll
        for (int c = 0; c < 2; ++c) {
          int vo = (dt * 16 + l16) * VSTR + 32 * c + hi * 8;
          v16bf bv = cat8(*(const v8bf*)&lv[vo], *(const v8bf*)&lv[vo + 16]);
          acc[dt] = __builtin_amdgcn_wmma_f32_16x16x32_bf16(false, pa[c], false, bv,
                                                            (short)0, acc[dt], false, false);
        }
    }
  }

  // ---- epilogue: normalize and store ----
#pragma unroll
  for (int r = 0; r < 8; ++r) {
    float inv = 1.0f / l[r];
    int row = qwave + r + hi * 8;
    float* op = Ob + (size_t)row * DHEAD + l16;
    op[0]  = acc[0][r] * inv;
    op[16] = acc[1][r] * inv;
    op[32] = acc[2][r] * inv;
    op[48] = acc[3][r] * inv;
  }
}

extern "C" void kernel_launch(void* const* d_in, const int* in_sizes, int n_in,
                              void* d_out, int out_size, void* d_ws, size_t ws_size,
                              hipStream_t stream) {
  const float* q = (const float*)d_in[0];
  const float* k = (const float*)d_in[1];
  const float* v = (const float*)d_in[2];
  float* o = (float*)d_out;
  int batches = in_sizes[0] / (T_SEQ * DHEAD);   // 8
  dim3 grid(T_SEQ / BLOCK_Q, batches);
  fa_causal_fwd<<<grid, 256, 0, stream>>>(q, k, v, o);
}
